// Attention_65223373357514
// MI455X (gfx1250) — compile-verified, matrix operand order checked by reading, not run on hardware
//
#include <hip/hip_runtime.h>

typedef __bf16 bf16_t;
typedef __attribute__((ext_vector_type(8)))  __bf16 v8bf;
typedef __attribute__((ext_vector_type(16))) __bf16 v16bf;
typedef __attribute__((ext_vector_type(8)))  float  v8f;
typedef __attribute__((ext_vector_type(4)))  unsigned int u32x4;
typedef __attribute__((ext_vector_type(4)))  int i32x4;
typedef __attribute__((ext_vector_type(8)))  int i32x8;

#define DM   2048   // DIM_MODEL
#define NH   16     // NUM_HEADS
#define NKV  4      // NUM_KV_HEADS
#define DH   128    // DIM_HEAD
#define BB   2      // batch
#define LQ   2048
#define LK   2048

// ---------------------------------------------------------------------------
// Feature detection: Tensor Data Mover builtin (arity differs per toolchain)
// ---------------------------------------------------------------------------
#if defined(__has_builtin)
#if __has_builtin(__builtin_amdgcn_tensor_load_to_lds)
#define HAVE_TDM 1
#endif
#if __has_builtin(__builtin_amdgcn_s_wait_tensorcnt)
#define HAVE_TENSORCNT 1
#endif
#endif

// ---------------------------------------------------------------------------
// WMMA helpers (CDNA5 wave32, v_wmma_f32_16x16x32_bf16)
// ---------------------------------------------------------------------------
__device__ inline v8f wmma_bf16(v16bf a, v16bf b, v8f c) {
  return __builtin_amdgcn_wmma_f32_16x16x32_bf16(
      /*neg_a=*/false, a, /*neg_b=*/false, b,
      /*c_mod=*/(short)0, c, /*reuse_a=*/false, /*reuse_b=*/false);
}

// Load a 16x32 bf16 A/B fragment from a K-major (row-major, ld = row stride
// in elements) tile.  Per ISA 7.12.2: lanes 0-15 hold M=lane with K chunks
// [0..7] (v0-3) and [16..23] (v4-7); lanes 16-31 hold M=lane-16 with K
// chunks [8..15] and [24..31].  -> two contiguous b128 loads per lane.
__device__ inline v16bf load_frag(const bf16_t* src, int ld) {
  const int lane = threadIdx.x & 31;
  const int m    = lane & 15;
  const int half = lane >> 4;
  const bf16_t* p = src + (size_t)m * ld + 8 * half;
  v8bf lo = *(const v8bf*)(p);        // K = 8*half .. 8*half+7
  v8bf hi = *(const v8bf*)(p + 16);   // K = 16+8*half .. +7
  v16bf r;
#pragma unroll
  for (int i = 0; i < 8; ++i) { r[i] = lo[i]; r[i + 8] = hi[i]; }
  return r;
}

// 16-lane row reductions (xor masks < 16 stay inside each half-wave)
__device__ inline float rowmax16(float v) {
#pragma unroll
  for (int m = 1; m < 16; m <<= 1) v = fmaxf(v, __shfl_xor(v, m, 32));
  return v;
}
__device__ inline float rowsum16(float v) {
#pragma unroll
  for (int m = 1; m < 16; m <<= 1) v += __shfl_xor(v, m, 32);
  return v;
}

// ---------------------------------------------------------------------------
// TDM: 2D tile DMA global -> LDS.  D# built per ISA 8.3/8.4 bit layout.
//   group0: [1:0]=count=1, [63:32]=lds_addr, [120:64]=global_addr, type=2
//   group1: data_size=1 (2 bytes), tensor_dim0/1, tile_dim0/1, dim0 stride
// ---------------------------------------------------------------------------
#if defined(HAVE_TDM)
__device__ inline void tdm_load_2d(unsigned lds_off, const void* gaddr,
                                   unsigned tile_d0, unsigned tile_d1,
                                   unsigned tensor_d0, unsigned tensor_d1,
                                   unsigned stride_elems) {
  const unsigned long long ga = (unsigned long long)(uintptr_t)gaddr;
  u32x4 g0;
  g0[0] = 1u;                                            // count=1, user D#
  g0[1] = lds_off;                                       // lds_addr (bytes)
  g0[2] = (unsigned)(ga & 0xffffffffu);                  // global_addr[31:0]
  g0[3] = (unsigned)((ga >> 32) & 0x01ffffffu) | (2u << 30);  // [56:32]|type=2
  i32x8 g1;
  g1[0] = 0x00010000;                                    // data_size=2B
  g1[1] = (int)((tensor_d0 & 0xffffu) << 16);            // tensor_dim0 lo16
  g1[2] = (int)(((tensor_d0 >> 16) & 0xffffu) |
                ((tensor_d1 & 0xffffu) << 16));          // dim0 hi | dim1 lo
  g1[3] = (int)(((tensor_d1 >> 16) & 0xffffu) |
                ((tile_d0 & 0xffffu) << 16));            // dim1 hi | tile0
  g1[4] = (int)(tile_d1 & 0xffffu);                      // tile_dim1
  g1[5] = (int)stride_elems;                             // dim0_stride lo32
  g1[6] = 0;
  g1[7] = 0;
  const i32x4 z4 = {0, 0, 0, 0};
#if __clang_major__ >= 23
  const i32x8 z8 = {0, 0, 0, 0, 0, 0, 0, 0};
  __builtin_amdgcn_tensor_load_to_lds(g0, g1, z4, z4, z8, 0);
#else
  __builtin_amdgcn_tensor_load_to_lds(g0, g1, z4, z4, 0);
#endif
}
#endif

// ---------------------------------------------------------------------------
// f32 -> bf16 convert with scale folding
// ---------------------------------------------------------------------------
__global__ void cvt_kernel(const float* __restrict__ src, bf16_t* __restrict__ dst,
                           int n, float scale) {
  int i = blockIdx.x * 256 + threadIdx.x;
  if (i < n) dst[i] = (bf16_t)(src[i] * scale);
}

// ---------------------------------------------------------------------------
// Tiled WMMA GEMM:  C(M,N) = A(M,Kd) * W(N,Kd)^T   (A, W both K-major bf16)
// Block: 256 threads = 8 waves; wave computes a 32x64 tile (2 A frags + 4
// shared B frags -> 8 WMMAs per K-step).  Block tile = 128(M) x 128(N).
// All dims exact multiples -> no bounds checks, EXEC all-ones around WMMA.
// Epilogue MODE: 0 = Q (b,head,lq,128) bf16     1 = K (b,kvh,lk,128) bf16
//                2 = V^T (b,kvh,128,lk) bf16    3 = f32 row-major (out proj)
// ---------------------------------------------------------------------------
template <int MODE>
__global__ __launch_bounds__(256)
void gemm_bf16_kernel(const bf16_t* __restrict__ A, const bf16_t* __restrict__ W,
                      int Kd, void* __restrict__ out) {
  const int wave = threadIdx.x >> 5;
  const int lane = threadIdx.x & 31;
  const int m0 = blockIdx.y * 128 + (wave & 3) * 32;
  const int n0 = blockIdx.x * 128 + (wave >> 2) * 64;

  v8f acc[2][4];
#pragma unroll
  for (int i = 0; i < 2; ++i)
#pragma unroll
    for (int j = 0; j < 4; ++j)
#pragma unroll
      for (int r = 0; r < 8; ++r) acc[i][j][r] = 0.0f;

  for (int kk = 0; kk < Kd; kk += 32) {
    v16bf a0 = load_frag(A + (size_t)m0 * Kd + kk, Kd);
    v16bf a1 = load_frag(A + (size_t)(m0 + 16) * Kd + kk, Kd);
#pragma unroll
    for (int j = 0; j < 4; ++j) {
      v16bf b = load_frag(W + (size_t)(n0 + 16 * j) * Kd + kk, Kd);
      acc[0][j] = wmma_bf16(a0, b, acc[0][j]);
      acc[1][j] = wmma_bf16(a1, b, acc[1][j]);
    }
  }

  const int half = lane >> 4, nl = lane & 15;
#pragma unroll
  for (int i = 0; i < 2; ++i) {
#pragma unroll
    for (int j = 0; j < 4; ++j) {
#pragma unroll
      for (int r = 0; r < 8; ++r) {
        const int row = m0 + 16 * i + r + 8 * half;  // global M (b*2048+seq)
        const int col = n0 + 16 * j + nl;            // global N
        const float v = acc[i][j][r];
        if (MODE == 3) {
          ((float*)out)[(size_t)row * DM + col] = v;
        } else {
          const int bb = row >> 11, sq = row & 2047;
          bf16_t* ob = (bf16_t*)out;
          if (MODE == 0) {  // Q: (b, head, lq, 128)
            const int h = col >> 7, d = col & 127;
            ob[((((size_t)bb * NH + h) * LQ) + sq) * DH + d] = (bf16_t)v;
          } else if (MODE == 1) {  // K: (b, kvh, lk, 128)
            const int h = col >> 7, d = col & 127;
            ob[((((size_t)bb * NKV + h) * LK) + sq) * DH + d] = (bf16_t)v;
          } else {  // MODE == 2, V^T: (b, kvh, 128, lk)
            const int h = col >> 7, d = col & 127;
            ob[(((size_t)bb * NKV + h) * DH + d) * LK + sq] = (bf16_t)v;
          }
        }
      }
    }
  }
}

// ---------------------------------------------------------------------------
// Flash attention, GQA-fused: one workgroup = 4 waves = the 4 heads of one
// KV group, all on the same 16-row Q block.  The 4 heads share K/V tiles:
// wave 0 drives a double-buffered Tensor-Data-Mover pipeline
// (tensor_load_to_lds for tile i+1, s_wait_tensorcnt<=2 for tile i, barrier
// publishes completion to the other waves), so each K/V tile is DMA'd once
// per group instead of once per head, and the DMA overlaps the 16 WMMAs +
// softmax of the current tile.  Falls back to direct global b128 loads when
// the TDM builtin is unavailable.
// ---------------------------------------------------------------------------
__global__ __launch_bounds__(128)
void attn_kernel(const bf16_t* __restrict__ Q, const bf16_t* __restrict__ Km,
                 const bf16_t* __restrict__ VT, const float* __restrict__ bias,
                 const unsigned char* __restrict__ mask,
                 bf16_t* __restrict__ attn_out) {
  __shared__ bf16_t plds[4][16 * 32];
#if defined(HAVE_TDM)
  __shared__ bf16_t kbuf[2][32 * DH];   // 2 x 8 KB: K tile, [key][d]
  __shared__ bf16_t vbuf[2][DH * 32];   // 2 x 8 KB: V^T tile, [d][key]
#endif

  const int wave = threadIdx.x >> 5;
  const int lane = threadIdx.x & 31;
  const int half = lane >> 4, nl = lane & 15;
  const int q0   = blockIdx.x * 16;
  const int kvh  = blockIdx.y;
  const int head = (kvh << 2) | wave;   // HEAD_GROUPS = 4
  const int b    = blockIdx.z;

  const bf16_t* Qp = Q  + ((((size_t)b * NH + head) * LQ) + q0) * DH;
  const bf16_t* Kp = Km + (((size_t)b * NKV + kvh) * LK) * DH;
  const bf16_t* Vp = VT + (((size_t)b * NKV + kvh) * DH) * LK;
  const float*  Bp = bias + ((size_t)head * LQ + q0) * LK;
  const unsigned char* Mp = mask + (((size_t)b * LQ) + q0) * LK;

  // Q fragments for the whole row block: 16 x 128 = 4 K-chunks of 32
  v16bf qa[4];
#pragma unroll
  for (int c = 0; c < 4; ++c) qa[c] = load_frag(Qp + 32 * c, DH);

  v8f o[8];
  float mrun[8], lrun[8];
#pragma unroll
  for (int r = 0; r < 8; ++r) {
    mrun[r] = -1e30f;
    lrun[r] = 0.0f;
#pragma unroll
    for (int j = 0; j < 8; ++j) o[j][r] = 0.0f;
  }

#if defined(HAVE_TDM)
  // Pipeline prologue: DMA tile 0 into buffer 0 (one wave owns TENSORcnt)
  if (wave == 0) {
    tdm_load_2d((unsigned)(uintptr_t)kbuf[0], Kp, DH, 32, DH, LK, DH);
    tdm_load_2d((unsigned)(uintptr_t)vbuf[0], Vp, 32, DH, LK, DH, LK);
  }
#endif

  for (int k0 = 0; k0 < LK; k0 += 32) {
#if defined(HAVE_TDM)
    const int cur = (k0 >> 5) & 1;
    __syncthreads();  // all waves done reading buf[cur^1] (prev tile, i-1)
    if (wave == 0) {
      if (k0 + 32 < LK) {  // prefetch tile i+1 into the other buffer
        tdm_load_2d((unsigned)(uintptr_t)kbuf[cur ^ 1],
                    Kp + (size_t)(k0 + 32) * DH, DH, 32, DH, LK, DH);
        tdm_load_2d((unsigned)(uintptr_t)vbuf[cur ^ 1],
                    Vp + (k0 + 32), 32, DH, LK, DH, LK);
#if defined(HAVE_TENSORCNT)
        __builtin_amdgcn_s_wait_tensorcnt(2);  // tile i done (in-order)
#endif
      } else {
#if defined(HAVE_TENSORCNT)
        __builtin_amdgcn_s_wait_tensorcnt(0);
#endif
      }
    }
    __syncthreads();  // tile i visible to all 4 head-waves
    const bf16_t* ksrc = kbuf[cur];
    const bf16_t* vsrc = vbuf[cur];
#endif
    // ---- scores: S(16x32) = Q(16x128) . K(32x128)^T --------------------
    v8f s[2];
#pragma unroll
    for (int t = 0; t < 2; ++t) {
#pragma unroll
      for (int r = 0; r < 8; ++r) s[t][r] = 0.0f;
#pragma unroll
      for (int c = 0; c < 4; ++c) {
#if defined(HAVE_TDM)
        v16bf kb = load_frag(ksrc + (size_t)(16 * t) * DH + 32 * c, DH);
#else
        v16bf kb = load_frag(Kp + (size_t)(k0 + 16 * t) * DH + 32 * c, DH);
#endif
        s[t] = wmma_bf16(qa[c], kb, s[t]);
      }
    }
    // ---- bias + mask (C layout: row = r + 8*half, col = nl) ------------
#pragma unroll
    for (int t = 0; t < 2; ++t) {
#pragma unroll
      for (int r = 0; r < 8; ++r) {
        const int row = r + 8 * half;
        const size_t idx = (size_t)row * LK + (k0 + 16 * t + nl);
        const float sc = s[t][r] + Bp[idx];
        s[t][r] = Mp[idx] ? sc : -1e30f;
      }
    }
    // ---- online softmax ------------------------------------------------
    float pn[2][8];
#pragma unroll
    for (int r = 0; r < 8; ++r) {
      float mx = rowmax16(fmaxf(s[0][r], s[1][r]));
      const float mnew = fmaxf(mrun[r], mx);
      const float resc = __expf(mrun[r] - mnew);
      const float p0 = __expf(s[0][r] - mnew);
      const float p1 = __expf(s[1][r] - mnew);
      pn[0][r] = p0; pn[1][r] = p1;
      lrun[r] = lrun[r] * resc + rowsum16(p0 + p1);
      mrun[r] = mnew;
#pragma unroll
      for (int j = 0; j < 8; ++j) o[j][r] *= resc;
    }
    // ---- permute P from C layout to A layout via per-wave LDS tile -----
#pragma unroll
    for (int t = 0; t < 2; ++t)
#pragma unroll
      for (int r = 0; r < 8; ++r)
        plds[wave][(r + 8 * half) * 32 + 16 * t + nl] = (bf16_t)pn[t][r];
    __syncthreads();
    v16bf pa = load_frag(plds[wave], 32);
    __syncthreads();
    // ---- O += P(16x32) . V(32x128) -------------------------------------
#pragma unroll
    for (int j = 0; j < 8; ++j) {
#if defined(HAVE_TDM)
      v16bf vb = load_frag(vsrc + (size_t)(16 * j) * 32, 32);
#else
      v16bf vb = load_frag(Vp + (size_t)(16 * j) * LK + k0, LK);
#endif
      o[j] = wmma_bf16(pa, vb, o[j]);
    }
  }

  // ---- normalize and store (b, lq, head*128 + d) bf16 -------------------
#pragma unroll
  for (int r = 0; r < 8; ++r) {
    const float inv = (lrun[r] > 0.0f) ? 1.0f / lrun[r] : 0.0f;
    const int row = q0 + r + 8 * half;
#pragma unroll
    for (int j = 0; j < 8; ++j) {
      attn_out[(((size_t)b * LQ + row) * (NH * DH)) + head * DH + 16 * j + nl] =
          (bf16_t)(o[j][r] * inv);
    }
  }
}

// ---------------------------------------------------------------------------
// Launcher
// ---------------------------------------------------------------------------
extern "C" void kernel_launch(void* const* d_in, const int* in_sizes, int n_in,
                              void* d_out, int out_size, void* d_ws, size_t ws_size,
                              hipStream_t stream) {
  const float* hq  = (const float*)d_in[0];
  const float* hkv = (const float*)d_in[1];
  const unsigned char* mask = (const unsigned char*)d_in[2];
  const float* pb  = (const float*)d_in[3];
  const float* Wq  = (const float*)d_in[4];
  const float* Wk  = (const float*)d_in[5];
  const float* Wv  = (const float*)d_in[6];
  const float* Wo  = (const float*)d_in[7];

  char* ws = (char*)d_ws;
  size_t off = 0;
  auto alloc = [&](size_t bytes) -> char* {
    char* p = ws + off;
    off += (bytes + 255) & ~(size_t)255;
    return p;
  };

  bf16_t* hq_b  = (bf16_t*)alloc((size_t)BB * LQ * DM * 2);
  bf16_t* hkv_b = (bf16_t*)alloc((size_t)BB * LK * DM * 2);
  bf16_t* wq_b  = (bf16_t*)alloc((size_t)NH  * DH * DM * 2);
  bf16_t* wk_b  = (bf16_t*)alloc((size_t)NKV * DH * DM * 2);
  bf16_t* wv_b  = (bf16_t*)alloc((size_t)NKV * DH * DM * 2);
  bf16_t* wo_b  = (bf16_t*)alloc((size_t)DM * NH * DH * 2);
  bf16_t* qh    = (bf16_t*)alloc((size_t)BB * NH  * LQ * DH * 2);
  bf16_t* kh    = (bf16_t*)alloc((size_t)BB * NKV * LK * DH * 2);
  bf16_t* vt    = (bf16_t*)alloc((size_t)BB * NKV * DH * LK * 2);
  bf16_t* attn  = (bf16_t*)alloc((size_t)BB * LQ * NH * DH * 2);

  const float inv4 = 0.2973017787506803f;  // 128^(-1/4), folded into Wq & Wk
  int n;
  n = BB * LQ * DM;  cvt_kernel<<<n / 256, 256, 0, stream>>>(hq,  hq_b,  n, 1.0f);
  n = BB * LK * DM;  cvt_kernel<<<n / 256, 256, 0, stream>>>(hkv, hkv_b, n, 1.0f);
  n = NH  * DH * DM; cvt_kernel<<<n / 256, 256, 0, stream>>>(Wq,  wq_b,  n, inv4);
  n = NKV * DH * DM; cvt_kernel<<<n / 256, 256, 0, stream>>>(Wk,  wk_b,  n, inv4);
  n = NKV * DH * DM; cvt_kernel<<<n / 256, 256, 0, stream>>>(Wv,  wv_b,  n, 1.0f);
  n = DM * NH * DH;  cvt_kernel<<<n / 256, 256, 0, stream>>>(Wo,  wo_b,  n, 1.0f);

  // Projections (block tile 128x128)
  gemm_bf16_kernel<0><<<dim3((NH * DH) / 128,  (BB * LQ) / 128), 256, 0, stream>>>(
      hq_b,  wq_b, DM, qh);
  gemm_bf16_kernel<1><<<dim3((NKV * DH) / 128, (BB * LK) / 128), 256, 0, stream>>>(
      hkv_b, wk_b, DM, kh);
  gemm_bf16_kernel<2><<<dim3((NKV * DH) / 128, (BB * LK) / 128), 256, 0, stream>>>(
      hkv_b, wv_b, DM, vt);

  // Attention core: one workgroup = 4 head-waves sharing TDM-staged K/V
  attn_kernel<<<dim3(LQ / 16, NKV, BB), 128, 0, stream>>>(qh, kh, vt, pb, mask, attn);

  // Output projection (f32 result straight into d_out)
  gemm_bf16_kernel<3><<<dim3(DM / 128, (BB * LQ) / 128), 256, 0, stream>>>(
      attn, wo_b, NH * DH, d_out);
}
